// AdditiveAttention_90486370992404
// MI455X (gfx1250) — compile-verified, matrix operand order checked by reading, not run on hardware
//
#include <hip/hip_runtime.h>
#include <math.h>

// ---------------- problem dims ----------------
#define QD 1024
#define HD 1024
#define KD 1024
#define BB 64
#define TT 2048
#define ROWS_TOTAL (BB * TT)

// ---------------- WMMA types ----------------
typedef __attribute__((ext_vector_type(16))) __bf16 v16bf;
typedef __attribute__((ext_vector_type(8)))  float  v8f;

union V16BF {
    v16bf v;
    uint4 u[2];
};

#if __has_builtin(__builtin_amdgcn_tanhf)
#define TANHF(x) __builtin_amdgcn_tanhf(x)
#else
#define TANHF(x) tanhf(x)
#endif

#if __has_builtin(__builtin_amdgcn_s_wait_asynccnt)
#define ASYNC_WAIT0() __builtin_amdgcn_s_wait_asynccnt(0)
#else
#define ASYNC_WAIT0() asm volatile("s_wait_asynccnt 0" ::: "memory")
#endif

// CDNA5 async global->LDS DMA copy, 16 bytes per lane (GV addressing mode).
// For LDS objects the low 32 bits of the generic address are the LDS byte offset.
__device__ __forceinline__ void async_copy_b128(void* lds_dst, const void* gsrc) {
    unsigned ldsoff = (unsigned)(unsigned long long)lds_dst;
    unsigned long long ga = (unsigned long long)gsrc;
    asm volatile("global_load_async_to_lds_b128 %0, %1, off"
                 :: "v"(ldsoff), "v"(ga) : "memory");
}

// float -> bf16 via native conversion (lowers to v_cvt_*bf16_f32), packed pair
__device__ __forceinline__ unsigned short f2bf_us(float a) {
    return __builtin_bit_cast(unsigned short, (__bf16)a);
}
__device__ __forceinline__ unsigned int pk2bf(float a, float b) {
    unsigned int lo = f2bf_us(a);
    unsigned int hi = f2bf_us(b);
    return lo | (hi << 16);
}

// ---------------- kernel 1: Wk (K x H, f32) -> Wk_t (H x K, bf16) ----------------
__global__ __launch_bounds__(256) void wk_transpose_kernel(const float* __restrict__ Wk,
                                                           unsigned short* __restrict__ wkt) {
    int gid = blockIdx.x * 256 + threadIdx.x;     // 1M threads
    int k = gid >> 10;
    int h = gid & 1023;
    wkt[h * 1024 + k] = f2bf_us(Wk[k * 1024 + h]);
}

// ---------------- kernel 2: q_proj = query @ Wq + bq + bk  (B x H, f32) ----------------
__global__ __launch_bounds__(256) void qproj_kernel(const float* __restrict__ q,
                                                    const float* __restrict__ Wq,
                                                    const float* __restrict__ bq,
                                                    const float* __restrict__ bk,
                                                    float* __restrict__ qp) {
    int gid = blockIdx.x * 256 + threadIdx.x;     // 65536 threads
    int b = gid >> 10;
    int h = gid & 1023;
    float s = bq[h] + bk[h];
    const float* qr = q + b * 1024;
    for (int k = 0; k < 1024; ++k)
        s += qr[k] * Wq[k * 1024 + h];
    qp[gid] = s;
}

// ---------------- kernel 3: fused  scores = tanh(keys@Wk + qproj) . Wv ----------------
// LDS layout: no pad; 32B-granule XOR swizzle (granule g of row r stored at g ^ (r&7)).
// Row blocks are 16-aligned so (row&7)==(lane&7): A and B share the swizzle term.
#define MT 128             // rows per workgroup (8 waves x 16 rows)

__global__ __launch_bounds__(256)
void scores_kernel(const float* __restrict__ keys,            // (B*T, K) f32
                   const unsigned short* __restrict__ wkt,    // (H, K) bf16
                   const float* __restrict__ qp,              // (B, H) f32  (bq+bk folded)
                   const float* __restrict__ Wv,              // (H)
                   float* __restrict__ scores)                // (B*T)
{
    __shared__ unsigned short Asm[MT * 1024];   // 262,144 B : keys tile (bf16, swizzled)
    __shared__ unsigned short Bsm[32 * 1024];   //  65,536 B : Wk_t 32-column chunk (swizzled)

    const int tid   = threadIdx.x;
    const int lane  = tid & 31;
    const int wave  = tid >> 5;            // 0..7 -> row block
    const int row0  = blockIdx.x * MT;
    const int b     = row0 / TT;           // MT divides TT -> one batch per WG

    // ---- stage A: keys[row0 .. row0+127][0..1023] -> bf16 in LDS (swizzled) ----
    {
        const float4* kp = (const float4*)(keys + (size_t)row0 * KD);
        #pragma unroll
        for (int j = 0; j < (MT * KD / 4) / 256; ++j) {    // 128 float4 per thread
            int i4 = tid + j * 256;
            int r  = i4 >> 8;              // 256 float4 per row
            int k4 = i4 & 255;
            float4 f = kp[(size_t)r * 256 + k4];
            // granule = k4>>2 ; within-granule uint2 slot = k4&3
            int idx = r * 1024 + (((k4 >> 2) ^ (r & 7)) << 4) + (k4 & 3) * 4;
            uint2 d;
            d.x = pk2bf(f.x, f.y);
            d.y = pk2bf(f.z, f.w);
            *(uint2*)&Asm[idx] = d;
        }
    }

    float part[8];
    #pragma unroll
    for (int i = 0; i < 8; ++i) part[i] = 0.f;

    const int m     = lane & 15;           // A row in block / B column in chunk
    const int khalf = lane >> 4;           // which 16-wide K slice this lane carries
    const int x7    = m & 7;               // shared swizzle term

    const unsigned short* aBase = &Asm[(wave * 16 + m) * 1024];
    const unsigned short* bBase = &Bsm[m * 1024];

    for (int nbp = 0; nbp < HD / 32; ++nbp) {              // 32 rounds of 32 columns
        __syncthreads();
        // ---- stage B chunk with CDNA5 async DMA: wkt rows nbp*32 .. +31, full K ----
        {
            const uint4* sp = (const uint4*)(wkt + (size_t)nbp * 32 * KD);
            #pragma unroll
            for (int j = 0; j < 4096 / 256; ++j) {     // 4096 uint4 total
                int i8 = tid + j * 256;
                int c  = i8 >> 7;                      // 128 uint4 per row
                int k8 = i8 & 127;
                int idx = c * 1024 + (((k8 >> 1) ^ (c & 7)) << 4) + (k8 & 1) * 8;
                async_copy_b128(&Bsm[idx], sp + c * 128 + k8);
            }
        }
        ASYNC_WAIT0();
        __syncthreads();

        v8f acc0 = {}, acc1 = {};
        #pragma unroll 4
        for (int kb = 0; kb < KD / 32; ++kb) {
            int off = (((kb * 2 + khalf) ^ x7) << 4);  // swizzled granule offset (ushorts)
            V16BF a, b0, b1;
            a.u[0]  = *(const uint4*)(aBase + off);
            a.u[1]  = *(const uint4*)(aBase + off + 8);
            b0.u[0] = *(const uint4*)(bBase + off);
            b0.u[1] = *(const uint4*)(bBase + off + 8);
            b1.u[0] = *(const uint4*)(bBase + 16384 + off);      // columns 16..31
            b1.u[1] = *(const uint4*)(bBase + 16384 + off + 8);
            acc0 = __builtin_amdgcn_wmma_f32_16x16x32_bf16(
                       false, a.v, false, b0.v, (short)0, acc0, false, false);
            acc1 = __builtin_amdgcn_wmma_f32_16x16x32_bf16(
                       false, a.v, false, b1.v, (short)0, acc1, false, false);
        }

        // epilogue: tanh + Wv-weighted accumulate for both 16-column chunks
        int n0 = nbp * 32 + m;
        float qv0  = qp[b * HD + n0];
        float qv1  = qp[b * HD + n0 + 16];
        float wv0  = Wv[n0];
        float wv1  = Wv[n0 + 16];
        #pragma unroll
        for (int i = 0; i < 8; ++i) {
            part[i] += TANHF(acc0[i] + qv0) * wv0;
            part[i] += TANHF(acc1[i] + qv1) * wv1;
        }
    }

    // reduce the 16 lanes sharing a row; lanes 0 / 16 hold rows i / 8+i
    #pragma unroll
    for (int i = 0; i < 8; ++i) {
        float s = part[i];
        s += __shfl_xor(s, 1, 32);
        s += __shfl_xor(s, 2, 32);
        s += __shfl_xor(s, 4, 32);
        s += __shfl_xor(s, 8, 32);
        if (m == 0)
            scores[row0 + wave * 16 + khalf * 8 + i] = s;
    }
}

// ---------------- kernel 4: softmax over T per batch; also zero context region ----------------
__global__ __launch_bounds__(256)
void softmax_kernel(const float* __restrict__ scores, float* __restrict__ out) {
    __shared__ float red[256];
    const int b   = blockIdx.x;
    const int tid = threadIdx.x;
    const float* s = scores + b * TT;

    float local[8];
    float mx = -1e30f;
    #pragma unroll
    for (int j = 0; j < 8; ++j) { local[j] = s[tid + j * 256]; mx = fmaxf(mx, local[j]); }
    red[tid] = mx; __syncthreads();
    for (int off = 128; off > 0; off >>= 1) {
        if (tid < off) red[tid] = fmaxf(red[tid], red[tid + off]);
        __syncthreads();
    }
    mx = red[0]; __syncthreads();

    float sum = 0.f;
    #pragma unroll
    for (int j = 0; j < 8; ++j) { local[j] = __expf(local[j] - mx); sum += local[j]; }
    red[tid] = sum; __syncthreads();
    for (int off = 128; off > 0; off >>= 1) {
        if (tid < off) red[tid] += red[tid + off];
        __syncthreads();
    }
    float inv = 1.f / red[0];

    float* attn = out + BB * HD;      // output layout: [context (B*1024) | attn (B*2048)]
    #pragma unroll
    for (int j = 0; j < 8; ++j)
        attn[b * TT + tid + j * 256] = local[j] * inv;

    // zero this batch's context slice before kernel 5 accumulates
    #pragma unroll
    for (int j = 0; j < 4; ++j)
        out[b * 1024 + tid + j * 256] = 0.f;
}

// ---------------- kernel 5: context = attn @ values ----------------
__global__ __launch_bounds__(256)
void context_kernel(const float* __restrict__ values,   // (B, T, K)
                    const float* __restrict__ attn,     // (B, T)
                    float* __restrict__ outc)           // (B, K)
{
    __shared__ float at[256];
    const int b   = blockIdx.x >> 3;
    const int tc  = blockIdx.x & 7;     // 8 t-chunks of 256
    const int tid = threadIdx.x;

    at[tid] = attn[b * TT + tc * 256 + tid];
    __syncthreads();

    float acc[4] = {0.f, 0.f, 0.f, 0.f};
    const float* vb = values + ((size_t)b * TT + tc * 256) * KD;
    for (int t = 0; t < 256; ++t) {
        float a = at[t];
        const float* vr = vb + (size_t)t * KD;
        #pragma unroll
        for (int j = 0; j < 4; ++j)
            acc[j] += a * vr[tid + j * 256];
    }
    #pragma unroll
    for (int j = 0; j < 4; ++j)
        atomicAdd(&outc[b * KD + tid + j * 256], acc[j]);
}

// ---------------- launcher ----------------
extern "C" void kernel_launch(void* const* d_in, const int* in_sizes, int n_in,
                              void* d_out, int out_size, void* d_ws, size_t ws_size,
                              hipStream_t stream) {
    const float* query  = (const float*)d_in[0];
    const float* keys   = (const float*)d_in[1];
    const float* values = (const float*)d_in[2];
    const float* Wq     = (const float*)d_in[3];
    const float* bq     = (const float*)d_in[4];
    const float* Wk     = (const float*)d_in[5];
    const float* bk     = (const float*)d_in[6];
    const float* Wv     = (const float*)d_in[7];
    // d_in[8] = bv : constant shift of scores -> softmax-invariant -> no effect on outputs

    float* out = (float*)d_out;
    char*  ws  = (char*)d_ws;

    unsigned short* wkt = (unsigned short*)ws;                            // 2 MB  bf16 Wk^T
    float* qp     = (float*)(ws + 2 * 1024 * 1024);                       // 256 KB q_proj(+bq+bk)
    float* scores = (float*)(ws + 2 * 1024 * 1024 + 256 * 1024);          // 512 KB scores

    wk_transpose_kernel<<<4096, 256, 0, stream>>>(Wk, wkt);
    qproj_kernel<<<256, 256, 0, stream>>>(query, Wq, bq, bk, qp);
    scores_kernel<<<ROWS_TOTAL / MT, 256, 0, stream>>>(keys, wkt, qp, Wv, scores);
    softmax_kernel<<<BB, 256, 0, stream>>>(scores, out);
    context_kernel<<<BB * 8, 256, 0, stream>>>(values, out + BB * HD, out);
}
